// FlashAttention_22952305230021
// MI455X (gfx1250) — matrix operand order repaired and checked by reading, not compiled, not run on hardware
//
#include <hip/hip_runtime.h>
#include <hip/hip_bf16.h>
#include <math.h>

// ---------------------------------------------------------------------------
// MI455X (gfx1250) flash-attention forward, bf16 WMMA + TDM-staged tiles
// throughout (QKV GEMM, attention K/V streaming, output projection).
// D=1024, H=16, Hd=64, L=4096, scale = 1/8.
// ---------------------------------------------------------------------------

#define SEQ     4096
#define DMODEL  1024
#define NHEADS  16
#define HDIM    64
#define SM_SCALE 0.125f

typedef __attribute__((ext_vector_type(16))) __bf16 v16bf;
typedef __attribute__((ext_vector_type(8)))  float  v8f;
typedef __attribute__((ext_vector_type(4)))  unsigned int u32x4;
typedef __attribute__((ext_vector_type(8)))  int          i32x8;
typedef __attribute__((ext_vector_type(4)))  int          i32x4;

union Frag16 { v16bf v; uint4 q[2]; };

__device__ __forceinline__ unsigned short f2bf(float f) {
  union { float f; unsigned u; } c; c.f = f;
  unsigned u = c.u + 0x7FFFu + ((c.u >> 16) & 1u);   // round-to-nearest-even
  return (unsigned short)(u >> 16);
}

__device__ __forceinline__ v8f wmma_bf16(v16bf a, v16bf b, v8f c) {
  return __builtin_amdgcn_wmma_f32_16x16x32_bf16(false, a, false, b,
                                                 (short)0, c, false, false);
}

// A fragment (16x32) from row-major bf16 [.. x stride], rows m0.., cols k0..
__device__ __forceinline__ v16bf load_a(const unsigned short* base, int stride,
                                        int m0, int k0, int lane) {
  const int m  = m0 + (lane & 15);
  const int hi = (lane >> 4) & 1;
  const unsigned short* p = base + (size_t)m * stride + k0;
  Frag16 f;
  f.q[0] = *(const uint4*)(p + 8 * hi);
  f.q[1] = *(const uint4*)(p + 16 + 8 * hi);
  return f.v;
}

// B fragment (32x16) from K-major bf16 storage Bt[N][K] (row n = contiguous K).
__device__ __forceinline__ v16bf load_b(const unsigned short* base, int stride,
                                        int n0, int k0, int lane) {
  const int n  = n0 + (lane & 15);
  const int hi = (lane >> 4) & 1;
  const unsigned short* p = base + (size_t)n * stride + k0 + 8 * hi;
  Frag16 f;
  f.q[0] = *(const uint4*)(p);
  f.q[1] = *(const uint4*)(p + 16);
  return f.v;
}

// ---------------------------------------------------------------------------
// Tensor Data Mover: async 2D tile load (bf16) Global -> LDS.
// ---------------------------------------------------------------------------
__device__ __forceinline__ unsigned lds_off32(const void* p) {
  return (unsigned)(unsigned long long)p;   // LDS aperture lives in addr[63:32]
}

__device__ __forceinline__ void tdm_load_2d_bf16(unsigned lds_addr,
                                                 const unsigned short* gptr,
                                                 unsigned tile_k,
                                                 unsigned tile_rows,
                                                 unsigned tensor_k,
                                                 unsigned tensor_rows,
                                                 unsigned stride_elems) {
  const unsigned long long ga = (unsigned long long)gptr;
  u32x4 g0;
  g0[0] = 1u;                                           // count=1, user mode
  g0[1] = lds_addr;                                     // lds_addr (bytes)
  g0[2] = (unsigned)(ga & 0xFFFFFFFFu);                 // global_addr[31:0]
  g0[3] = (unsigned)((ga >> 32) & 0x1FFFFFFu) | (2u << 30);  // [56:32] | type=2
  i32x8 g1;
  g1[0] = (int)(1u << 16);                              // data_size=1 -> 2 bytes
  g1[1] = (int)((tensor_k & 0xFFFFu) << 16);            // tensor_dim0[15:0]
  g1[2] = (int)(((tensor_k >> 16) & 0xFFFFu) | ((tensor_rows & 0xFFFFu) << 16));
  g1[3] = (int)(((tensor_rows >> 16) & 0xFFFFu) | ((tile_k & 0xFFFFu) << 16));
  g1[4] = (int)(tile_rows & 0xFFFFu);                   // tile_dim1; tile_dim2=0
  g1[5] = (int)stride_elems;                            // dim0 stride [31:0]
  g1[6] = 0;
  g1[7] = 0;
  i32x4 z4 = {0, 0, 0, 0};
#if defined(__clang_major__) && (__clang_major__ >= 23)
  i32x8 z8 = {0, 0, 0, 0, 0, 0, 0, 0};
  __builtin_amdgcn_tensor_load_to_lds(g0, g1, z4, z4, z8, 0);
#else
  __builtin_amdgcn_tensor_load_to_lds(g0, g1, z4, z4, 0);
#endif
}

// ---------------------------------------------------------------------------
// fp32 -> bf16 cast (grid-stride)
// ---------------------------------------------------------------------------
__global__ void cast_bf16_kernel(const float* __restrict__ in,
                                 unsigned short* __restrict__ out, int n) {
  int i  = blockIdx.x * blockDim.x + threadIdx.x;
  int st = gridDim.x * blockDim.x;
  for (; i < n; i += st) out[i] = f2bf(in[i]);
}

// ---------------------------------------------------------------------------
// Transpose + cast: Wt[n][k] = bf16(W[k][n]).  32x32 LDS tiles, block = 256.
// ---------------------------------------------------------------------------
__global__ void transpose_cast_kernel(const float* __restrict__ W,
                                      unsigned short* __restrict__ Wt,
                                      int K, int N) {
  __shared__ float tile[32][33];
  const int k0 = blockIdx.x * 32;
  const int n0 = blockIdx.y * 32;
  const int tx = threadIdx.x & 31;
  const int ty = threadIdx.x >> 5;
  #pragma unroll
  for (int r = ty; r < 32; r += 8)
    tile[r][tx] = W[(size_t)(k0 + r) * N + (n0 + tx)];
  __syncthreads();
  #pragma unroll
  for (int r = ty; r < 32; r += 8)
    Wt[(size_t)(n0 + r) * K + (k0 + tx)] = f2bf(tile[tx][r]);
}

// ---------------------------------------------------------------------------
// TDM-staged GEMM body. Block = 256 threads (8 waves), tile 128(M) x 64(N),
// k-step 32, double-buffered LDS fed by the Tensor Data Mover.
// Single barrier per k-step: the top barrier of step i proves all reads of
// buffer ((i-1)&1) finished, and the stage issued in step i writes that buffer.
// ---------------------------------------------------------------------------
#define GEMM_STAGE(bufidx, kk)                                                  \
  do {                                                                          \
    tdm_load_2d_bf16(lds_off32(&Asm[bufidx][0][0]), A + (size_t)mblk * K + (kk),\
                     32, 128, K, M, K);                                         \
    tdm_load_2d_bf16(lds_off32(&Bsm[bufidx][0][0]), Bt + (size_t)n0 * K + (kk), \
                     32, 64, K, N, K);                                          \
  } while (0)

#define GEMM_BODY(EPILOGUE)                                                     \
  __shared__ __align__(16) unsigned short Asm[2][128][32];                      \
  __shared__ __align__(16) unsigned short Bsm[2][64][32];                       \
  const int lane = threadIdx.x & 31;                                            \
  const int wave = threadIdx.x >> 5;                                            \
  const int hi   = (lane >> 4) & 1;                                             \
  const int mblk = blockIdx.y * 128;                                            \
  const int n0   = blockIdx.x * 64;                                             \
  const int mloc = wave * 16;                                                   \
  v8f acc[4];                                                                   \
  _Pragma("unroll")                                                             \
  for (int g = 0; g < 4; ++g)                                                   \
    _Pragma("unroll")                                                           \
    for (int i = 0; i < 8; ++i) acc[g][i] = 0.0f;                               \
  if (wave == 0) GEMM_STAGE(0, 0);                                              \
  for (int k0 = 0; k0 < K; k0 += 32) {                                          \
    const int cur = (k0 >> 5) & 1;                                              \
    __builtin_amdgcn_s_wait_tensorcnt(0);                                       \
    __syncthreads();                         /* tile 'cur' ready for all */     \
    if ((k0 + 32 < K) && wave == 0) GEMM_STAGE(cur ^ 1, k0 + 32);               \
    v16bf a = load_a(&Asm[cur][0][0], 32, mloc, 0, lane);                       \
    _Pragma("unroll")                                                           \
    for (int g = 0; g < 4; ++g) {                                               \
      v16bf b = load_b(&Bsm[cur][0][0], 32, g * 16, 0, lane);                   \
      acc[g] = wmma_bf16(a, b, acc[g]);                                         \
    }                                                                           \
  }                                                                             \
  EPILOGUE

// QKV GEMM: X[4096x1024] @ Wqkv_t -> Q,K rows in QK[4096][2048], V into
// Vt[1024][4096] (transposed per-dim rows for the P.V B fragments).
__global__ void qkv_gemm_kernel(const unsigned short* __restrict__ A,
                                const unsigned short* __restrict__ Bt,
                                unsigned short* __restrict__ QK,
                                unsigned short* __restrict__ Vt) {
  const int M = SEQ, N = 3 * DMODEL, K = DMODEL;
  GEMM_BODY({
    _Pragma("unroll")
    for (int g = 0; g < 4; ++g) {
      const int col = n0 + g * 16 + (lane & 15);
      _Pragma("unroll")
      for (int r = 0; r < 8; ++r) {
        const int row = mblk + mloc + r + 8 * hi;
        const unsigned short h = f2bf(acc[g][r]);
        if (col < 2 * DMODEL)
          QK[(size_t)row * (2 * DMODEL) + col] = h;
        else
          Vt[(size_t)(col - 2 * DMODEL) * SEQ + row] = h;
      }
    }
  })
}

// Output projection: O[4096x1024] @ Wo_t + b_o -> fp32 out.
__global__ void out_gemm_kernel(const unsigned short* __restrict__ A,
                                const unsigned short* __restrict__ Bt,
                                const float* __restrict__ bias,
                                float* __restrict__ out) {
  const int M = SEQ, N = DMODEL, K = DMODEL;
  GEMM_BODY({
    _Pragma("unroll")
    for (int g = 0; g < 4; ++g) {
      const int col = n0 + g * 16 + (lane & 15);
      const float bv = bias[col];
      _Pragma("unroll")
      for (int r = 0; r < 8; ++r) {
        const int row = mblk + mloc + r + 8 * hi;
        out[(size_t)row * DMODEL + col] = acc[g][r] + bv;
      }
    }
  })
}

// ---------------------------------------------------------------------------
// Flash attention: block = 128 (4 waves), all waves share one head, each wave
// owns one 16-query tile. K/V tiles for each 32-key block are staged into
// double-buffered LDS by the TDM (shared across the 4 waves, fills overlap
// the softmax+WMMA compute of the previous block).
// ---------------------------------------------------------------------------
#define ATTN_STAGE(bufidx, jj)                                                  \
  do {                                                                          \
    tdm_load_2d_bf16(lds_off32(&Ksm[bufidx][0][0]),                             \
                     Kg + (size_t)(jj) * (2 * DMODEL), 64, 32, 64, 32,          \
                     2 * DMODEL);                                               \
    tdm_load_2d_bf16(lds_off32(&Vsm[bufidx][0][0]),                             \
                     Vt + (size_t)qcol * SEQ + (jj), 32, 64, 32, 64, SEQ);      \
  } while (0)

__global__ void attn_kernel(const unsigned short* __restrict__ QK,
                            const unsigned short* __restrict__ Vt,
                            unsigned short* __restrict__ O) {
  const int lane = threadIdx.x & 31;
  const int wave = threadIdx.x >> 5;
  const int hi   = (lane >> 4) & 1;
  const int head = blockIdx.y;
  const int q0   = (blockIdx.x * 4 + wave) * 16;
  const int qcol = head * HDIM;

  __shared__ __align__(16) unsigned short Ksm[2][32][64];   // [key][dim]
  __shared__ __align__(16) unsigned short Vsm[2][64][32];   // [dim][key]
  __shared__ __align__(16) unsigned short ptile[4][16 * 32];
  unsigned short* lds = ptile[wave];

  const v16bf aq0 = load_a(QK, 2 * DMODEL, q0, qcol + 0,  lane);
  const v16bf aq1 = load_a(QK, 2 * DMODEL, q0, qcol + 32, lane);

  v8f  acc[4];
  float mi[8], li[8];
  #pragma unroll
  for (int g = 0; g < 4; ++g)
    #pragma unroll
    for (int i = 0; i < 8; ++i) acc[g][i] = 0.0f;
  #pragma unroll
  for (int r = 0; r < 8; ++r) { mi[r] = -1e30f; li[r] = 0.0f; }

  const unsigned short* Kg = QK + DMODEL + qcol;   // key rows, head's K columns

  if (wave == 0) ATTN_STAGE(0, 0);

  for (int j0 = 0; j0 < SEQ; j0 += 32) {
    const int cur = (j0 >> 5) & 1;
    __builtin_amdgcn_s_wait_tensorcnt(0);
    __syncthreads();                            // K/V tile 'cur' ready
    if ((j0 + 32 < SEQ) && wave == 0) ATTN_STAGE(cur ^ 1, j0 + 32);

    // ---- S = Q . K^T from LDS tile ---------------------------------------
    v8f s0, s1;
    #pragma unroll
    for (int i = 0; i < 8; ++i) { s0[i] = 0.0f; s1[i] = 0.0f; }
    s0 = wmma_bf16(aq0, load_b(&Ksm[cur][0][0], 64, 0,  0,  lane), s0);
    s0 = wmma_bf16(aq1, load_b(&Ksm[cur][0][0], 64, 0,  32, lane), s0);
    s1 = wmma_bf16(aq0, load_b(&Ksm[cur][0][0], 64, 16, 0,  lane), s1);
    s1 = wmma_bf16(aq1, load_b(&Ksm[cur][0][0], 64, 16, 32, lane), s1);

    // ---- online softmax --------------------------------------------------
    float alpha[8];
    #pragma unroll
    for (int r = 0; r < 8; ++r) {
      const float a0 = s0[r] * SM_SCALE;
      const float a1 = s1[r] * SM_SCALE;
      float bm = fmaxf(a0, a1);
      #pragma unroll
      for (int m = 1; m < 16; m <<= 1) bm = fmaxf(bm, __shfl_xor(bm, m, 32));
      const float mnew = fmaxf(mi[r], bm);
      const float al   = __expf(mi[r] - mnew);
      const float p0   = __expf(a0 - mnew);
      const float p1   = __expf(a1 - mnew);
      float rs = p0 + p1;
      #pragma unroll
      for (int m = 1; m < 16; m <<= 1) rs += __shfl_xor(rs, m, 32);
      li[r] = li[r] * al + rs;
      mi[r] = mnew;
      alpha[r] = al;
      lds[(r + 8 * hi) * 32 + (lane & 15)]      = f2bf(p0);
      lds[(r + 8 * hi) * 32 + (lane & 15) + 16] = f2bf(p1);
    }
    #pragma unroll
    for (int g = 0; g < 4; ++g)
      #pragma unroll
      for (int r = 0; r < 8; ++r) acc[g][r] *= alpha[r];

    asm volatile("s_wait_dscnt 0" ::: "memory");   // wave-private P tile

    // ---- O += P . V from LDS tile ----------------------------------------
    const v16bf pa = load_a(lds, 32, 0, 0, lane);
    #pragma unroll
    for (int g = 0; g < 4; ++g) {
      v16bf bv = load_b(&Vsm[cur][0][0], 32, g * 16, 0, lane);
      acc[g] = wmma_bf16(pa, bv, acc[g]);
    }
  }

  float inv[8];
  #pragma unroll
  for (int r = 0; r < 8; ++r) inv[r] = 1.0f / li[r];
  #pragma unroll
  for (int g = 0; g < 4; ++g) {
    const int col = qcol + g * 16 + (lane & 15);
    #pragma unroll
    for (int r = 0; r < 8; ++r) {
      const int row = q0 + r + 8 * hi;
      O[(size_t)row * DMODEL + col] = f2bf(acc[g][r] * inv[r]);
    }
  }
}

// ---------------------------------------------------------------------------
// Host-side orchestration
// ---------------------------------------------------------------------------
extern "C" void kernel_launch(void* const* d_in, const int* in_sizes, int n_in,
                              void* d_out, int out_size, void* d_ws, size_t ws_size,
                              hipStream_t stream) {
  const float* x    = (const float*)d_in[0];   // [4096,1024]
  const float* Wqkv = (const float*)d_in[1];   // [1024,3072]
  const float* Wo   = (const float*)d_in[2];   // [1024,1024]
  const float* bo   = (const float*)d_in[3];   // [1024]
  float* out = (float*)d_out;

  unsigned char* ws = (unsigned char*)d_ws;
  size_t off = 0;
  unsigned short* xb  = (unsigned short*)(ws + off); off += (size_t)SEQ * DMODEL * 2;
  unsigned short* wqt = (unsigned short*)(ws + off); off += (size_t)3 * DMODEL * DMODEL * 2;
  unsigned short* qk  = (unsigned short*)(ws + off); off += (size_t)SEQ * 2 * DMODEL * 2;
  unsigned short* vt  = (unsigned short*)(ws + off); off += (size_t)DMODEL * SEQ * 2;
  unsigned short* ob  = (unsigned short*)(ws + off); off += (size_t)SEQ * DMODEL * 2;
  unsigned short* wot = (unsigned short*)(ws + off); off += (size_t)DMODEL * DMODEL * 2;

  cast_bf16_kernel<<<512, 256, 0, stream>>>(x, xb, SEQ * DMODEL);
  transpose_cast_kernel<<<dim3(DMODEL / 32, 3 * DMODEL / 32), 256, 0, stream>>>(
      Wqkv, wqt, DMODEL, 3 * DMODEL);
  transpose_cast_kernel<<<dim3(DMODEL / 32, DMODEL / 32), 256, 0, stream>>>(
      Wo, wot, DMODEL, DMODEL);
  qkv_gemm_kernel<<<dim3(3 * DMODEL / 64, SEQ / 128), 256, 0, stream>>>(xb, wqt, qk, vt);
  attn_kernel<<<dim3(SEQ / 64, NHEADS), 128, 0, stream>>>(qk, vt, ob);
  out_gemm_kernel<<<dim3(DMODEL / 64, SEQ / 128), 256, 0, stream>>>(ob, wot, bo, out);
}
